// AdaptiveAttention_21139829031335
// MI455X (gfx1250) — matrix operand order repaired and checked by reading, not compiled, not run on hardware
//
#include <hip/hip_runtime.h>
#include <hip/hip_bf16.h>
#include <stdint.h>

typedef __attribute__((ext_vector_type(16))) __bf16 v16bf;
typedef __attribute__((ext_vector_type(8)))  float  v8f;

#define NUM_HEADS 8
#define HEAD_DIM  64

static __device__ __forceinline__ unsigned short f32_to_bf16(float f) {
    union { float f; uint32_t u; } v; v.f = f;
    uint32_t u = v.u;
    uint32_t r = u + 0x7FFFu + ((u >> 16) & 1u);   // round-to-nearest-even
    return (unsigned short)(r >> 16);
}
static __device__ __forceinline__ float bf16_to_f32(unsigned short h) {
    union { uint32_t u; float f; } v; v.u = ((uint32_t)h) << 16; return v.f;
}

// ---- WMMA operand swizzles (ushort-element offsets) ----------------------
// A-matrix 16x32 bf16 fragment layout (ISA 7.12.2): lane L<16 holds M=L,
// K in {0..7,16..23}; lane L>=16 holds M=L-16, K in {8..15,24..31}.
// One 32-lane x 32B block per (16-row, 32-K) tile; lane's 16 values contiguous.
static __device__ __forceinline__ size_t a_swz_idx(int m, int k, int Ktot) {
    int mt = m >> 4, lr = m & 15;
    int kt = k >> 5, kl = k & 31;
    int k2 = kl & 15, half2 = kl >> 4;
    int laneHi = (k2 >> 3) & 1;
    int lane = lr + (laneHi << 4);
    int e = (half2 << 3) + (k2 & 7);
    size_t blk = (size_t)mt * (size_t)(Ktot >> 5) + (size_t)kt;
    return blk * 512 + (size_t)lane * 16 + (size_t)e;
}
// B-matrix 32x16 bf16 fragment layout: lanes 0-15 = columns, K {0..7,16..23};
// lanes 16-31 same columns, K {8..15,24..31}.
static __device__ __forceinline__ size_t b_swz_idx(int k, int n, int Ntot) {
    int kt = k >> 5, kl = k & 31;
    int nt = n >> 4, nl = n & 15;
    int lane = nl + (((kl >> 3) & 1) << 4);
    int e = ((kl >> 4) << 3) + (kl & 7);
    size_t blk = (size_t)kt * (size_t)(Ntot >> 4) + (size_t)nt;
    return blk * 512 + (size_t)lane * 16 + (size_t)e;
}

// ---- Kernel 1: x + linear-interp(pos), convert bf16, A-swizzled store ----
__global__ void prep_x_kernel(const float* __restrict__ x,
                              const float* __restrict__ pos32,
                              const int* __restrict__ resolution,
                              unsigned short* __restrict__ xa,
                              int Nn, int Cn, int posLen) {
    size_t tid = (size_t)blockIdx.x * blockDim.x + threadIdx.x;
    int c = (int)(tid % (size_t)Cn);
    size_t mn = tid / (size_t)Cn;
    int n = (int)(mn % (size_t)Nn);
    float val = x[tid];
    int r = resolution[0];
    int target = r * r * r;
    if (target == Nn) {
        float scale = (float)posLen / (float)target;
        float coord = ((float)n + 0.5f) * scale - 0.5f;
        coord = fminf(fmaxf(coord, 0.0f), (float)(posLen - 1));
        int i0 = (int)floorf(coord);
        int i1 = min(i0 + 1, posLen - 1);
        float w = coord - (float)i0;
        val += pos32[(size_t)i0 * Cn + c] * (1.0f - w)
             + pos32[(size_t)i1 * Cn + c] * w;
    }
    xa[a_swz_idx((int)mn, c, Cn)] = f32_to_bf16(val);
}

// ---- Kernel 2: weight f32 -> bf16 in B-fragment layout -------------------
__global__ void conv_w_kernel(const float* __restrict__ w,
                              unsigned short* __restrict__ wb,
                              int Kd, int Nd) {
    size_t tid = (size_t)blockIdx.x * blockDim.x + threadIdx.x;
    if (tid >= (size_t)Kd * Nd) return;
    int n = (int)(tid % (size_t)Nd);
    int k = (int)(tid / (size_t)Nd);
    wb[b_swz_idx(k, n, Nd)] = f32_to_bf16(w[tid]);
}

// ---- Kernel 3/5: LDS-staged bf16 WMMA GEMM -------------------------------
// Block tile 128x128 (8 waves; each wave 64x32 = 8 WMMAs/K-step).
// Operands pre-swizzled in global so global->LDS copy is contiguous uint4 and
// LDS fragment reads are two ds_load_b128 per lane.
template <bool OUT_BF16>
__global__ void __launch_bounds__(256)
gemm_wmma_kernel(const unsigned short* __restrict__ A,   // A-swizzled M x K
                 const unsigned short* __restrict__ Bm,  // B-swizzled K x N
                 const float* __restrict__ bias,         // length N
                 void* __restrict__ out,                 // row-major M x N
                 int M, int Nd, int Kd) {
    __shared__ __align__(16) unsigned short ldsA[2][8 * 512];  // 8 A-blocks (16 rows x 32 K)
    __shared__ __align__(16) unsigned short ldsB[2][8 * 512];  // 8 B-blocks (32 K x 16 cols)

    const int lane = threadIdx.x;
    const int w    = threadIdx.y;
    const int tid  = w * 32 + lane;

    const int nBlkN = Nd >> 7;                 // 128-wide tiles in N
    const int bm = blockIdx.x / nBlkN;         // block tile row
    const int bn = blockIdx.x % nBlkN;         // block tile col
    const int bm8 = bm * 8;                    // first A 16-row block
    const int bn8 = bn * 8;                    // first B 16-col block

    const int nKt = Kd >> 5;
    const int nNt = Nd >> 4;
    const uint4* A4 = (const uint4*)A;         // 64 uint4 per 1KB fragment block
    const uint4* B4 = (const uint4*)Bm;

    // wave -> 64x32 sub-tile: 2x4 wave grid
    const int wm = w >> 2;                     // 0..1 (M offset wm*64)
    const int wn = w & 3;                      // 0..3 (N offset wn*32)

    v8f acc[4][2];
    #pragma unroll
    for (int i = 0; i < 4; ++i)
        #pragma unroll
        for (int j = 0; j < 2; ++j)
            acc[i][j] = v8f{};

    auto copy_tile = [&](int p, int kt) {
        uint4* dA = (uint4*)&ldsA[p][0];
        uint4* dB = (uint4*)&ldsB[p][0];
        #pragma unroll
        for (int r = 0; r < 2; ++r) {
            int idx = tid + r * 256;           // 0..511 uint4
            int blk = idx >> 6, off = idx & 63;
            dA[idx] = A4[((size_t)(bm8 + blk) * nKt + kt) * 64 + off];
            dB[idx] = B4[((size_t)kt * nNt + (bn8 + blk)) * 64 + off];
        }
    };

    copy_tile(0, 0);
    __syncthreads();

    for (int kt = 0; kt < nKt; ++kt) {
        int p = kt & 1;
        if (kt + 1 < nKt) copy_tile(p ^ 1, kt + 1);

        const uint4* la = (const uint4*)&ldsA[p][0] + (size_t)(wm * 4) * 64;
        const uint4* lb = (const uint4*)&ldsB[p][0] + (size_t)(wn * 2) * 64;
        union U { uint4 q[2]; v16bf v; };
        U b0, b1;
        b0.q[0] = lb[lane * 2];      b0.q[1] = lb[lane * 2 + 1];
        b1.q[0] = lb[64 + lane * 2]; b1.q[1] = lb[64 + lane * 2 + 1];
        #pragma unroll
        for (int i = 0; i < 4; ++i) {
            U a;
            a.q[0] = la[i * 64 + lane * 2];
            a.q[1] = la[i * 64 + lane * 2 + 1];
            acc[i][0] = __builtin_amdgcn_wmma_f32_16x16x32_bf16(false, a.v, false, b0.v, (short)0, acc[i][0], false, false);
            acc[i][1] = __builtin_amdgcn_wmma_f32_16x16x32_bf16(false, a.v, false, b1.v, (short)0, acc[i][1], false, false);
        }
        __syncthreads();
    }

    // D layout: lanes 0-15 col=lane, rows r; lanes 16-31 col=lane-16, rows r+8
    const int colLo  = lane & 15;
    const int rowOff = (lane >> 4) * 8;
    #pragma unroll
    for (int i = 0; i < 4; ++i) {
        #pragma unroll
        for (int j = 0; j < 2; ++j) {
            int col = bn * 128 + wn * 32 + j * 16 + colLo;
            float bv = bias[col];
            int rowBase = bm * 128 + wm * 64 + i * 16 + rowOff;
            #pragma unroll
            for (int r2 = 0; r2 < 8; ++r2) {
                float v = acc[i][j][r2] + bv;
                size_t idx = (size_t)(rowBase + r2) * Nd + col;
                if (OUT_BF16) ((unsigned short*)out)[idx] = f32_to_bf16(v);
                else          ((float*)out)[idx] = v;
            }
        }
    }
}

// ---- Kernel 4: per-token 8x8 cross-head attention ------------------------
// One wave per token. Output scrambled per reference transpose+reshape:
// row n2 = h*(N/8) + n/8, col c2 = (n%8)*64 + d, stored A-swizzled for GEMM2.
__global__ void attn_kernel(const unsigned short* __restrict__ qkv, // (B*N) x 1536 bf16
                            unsigned short* __restrict__ outA,      // A-swizzled (B*N) x 512
                            int Nn, int Cn) {
    __shared__ float sm[8][3 * 512 + 64];   // q,k,v,S per wave
    int lane = threadIdx.x;
    int w = threadIdx.y;
    int t = blockIdx.x * 8 + w;             // global token id (grid divides exactly)

    float* q = sm[w];
    float* k = q + 512;
    float* v = k + 512;
    float* S = v + 512;

    // load row (1536 bf16 = 192 uint4), convert to f32 in LDS
    const uint4* row = (const uint4*)(qkv + (size_t)t * (3 * (size_t)Cn));
    #pragma unroll
    for (int i = 0; i < 6; ++i) {
        uint4 d = row[lane * 6 + i];
        const unsigned short* hs = (const unsigned short*)&d;
        int base = lane * 48 + i * 8;
        #pragma unroll
        for (int e = 0; e < 8; ++e) q[base + e] = bf16_to_f32(hs[e]);
    }
    __builtin_amdgcn_s_barrier_signal(-1);
    __builtin_amdgcn_s_barrier_wait(-1);

    // S[i][j] = q_i . k_j / sqrt(64); 64 pairs over 32 lanes
    const float scale = 0.125f;
    #pragma unroll
    for (int pp = 0; pp < 2; ++pp) {
        int p = lane + pp * 32;
        int i = p >> 3, j = p & 7;
        float s = 0.0f;
        for (int d2 = 0; d2 < HEAD_DIM; ++d2) s += q[i * 64 + d2] * k[j * 64 + d2];
        S[p] = s * scale;
    }
    __builtin_amdgcn_s_barrier_signal(-1);
    __builtin_amdgcn_s_barrier_wait(-1);

    // lane -> (head i = lane/4, dim chunk of 16)
    int i = lane >> 2;
    int dbase = (lane & 3) << 4;
    float Pr[8];
    float mx = -3.0e38f;
    #pragma unroll
    for (int j = 0; j < 8; ++j) mx = fmaxf(mx, S[i * 8 + j]);
    float sum = 0.0f;
    #pragma unroll
    for (int j = 0; j < 8; ++j) { Pr[j] = __expf(S[i * 8 + j] - mx); sum += Pr[j]; }
    float inv = 1.0f / sum;

    int b = t / Nn, n = t % Nn;
    int n2 = i * (Nn >> 3) + (n >> 3);
    int m2 = b * Nn + n2;
    int c2base = (n & 7) * 64 + dbase;
    #pragma unroll
    for (int dd = 0; dd < 16; ++dd) {
        float o = 0.0f;
        #pragma unroll
        for (int j = 0; j < 8; ++j) o += Pr[j] * v[j * 64 + dbase + dd];
        outA[a_swz_idx(m2, c2base + dd, Cn)] = f32_to_bf16(o * inv);
    }
}

extern "C" void kernel_launch(void* const* d_in, const int* in_sizes, int n_in,
                              void* d_out, int out_size, void* d_ws, size_t ws_size,
                              hipStream_t stream) {
    const float* x      = (const float*)d_in[0];
    const float* pos32  = (const float*)d_in[1];
    const float* w_qkv  = (const float*)d_in[2];
    const float* b_qkv  = (const float*)d_in[3];
    const float* w_proj = (const float*)d_in[4];
    const float* b_proj = (const float*)d_in[5];
    const int*   reso   = (const int*)d_in[6];
    float* out = (float*)d_out;

    const int C = in_sizes[3] / 3;          // 512
    const int BN = in_sizes[0] / C;         // 32768
    const int B = 8;
    const int N = BN / B;                   // 4096
    const int C3 = 3 * C;                   // 1536
    const int posLen = in_sizes[1] / C;     // 32768

    // workspace layout (bytes, 256-aligned)
    auto align256 = [](size_t s) { return (s + 255) & ~(size_t)255; };
    char* ws = (char*)d_ws;
    size_t off = 0;
    unsigned short* xa = (unsigned short*)(ws + off); off += align256((size_t)BN * C * 2);
    unsigned short* wqkvB = (unsigned short*)(ws + off); off += align256((size_t)C * C3 * 2);
    unsigned short* wprojB = (unsigned short*)(ws + off); off += align256((size_t)C * C * 2);
    unsigned short* qkv = (unsigned short*)(ws + off); off += align256((size_t)BN * C3 * 2);
    unsigned short* attnA = (unsigned short*)(ws + off); off += align256((size_t)BN * C * 2);
    (void)ws_size; (void)n_in; (void)out_size;

    // 1) x + interp(pos) -> bf16 A-swizzled
    {
        size_t total = (size_t)BN * C;
        prep_x_kernel<<<dim3((unsigned)(total / 256)), dim3(256), 0, stream>>>(
            x, pos32, reso, xa, N, C, posLen);
    }
    // 2) weights -> bf16 B-swizzled
    {
        size_t t1 = (size_t)C * C3;
        conv_w_kernel<<<dim3((unsigned)((t1 + 255) / 256)), dim3(256), 0, stream>>>(w_qkv, wqkvB, C, C3);
        size_t t2 = (size_t)C * C;
        conv_w_kernel<<<dim3((unsigned)((t2 + 255) / 256)), dim3(256), 0, stream>>>(w_proj, wprojB, C, C);
    }
    // 3) QKV GEMM (bf16 out, bias fused): 128x128 block tiles
    {
        int blocks = (BN / 128) * (C3 / 128);
        gemm_wmma_kernel<true><<<dim3(blocks), dim3(32, 8), 0, stream>>>(
            xa, wqkvB, b_qkv, (void*)qkv, BN, C3, C);
    }
    // 4) per-token cross-head attention -> bf16 A-swizzled (scrambled rows)
    {
        attn_kernel<<<dim3(BN / 8), dim3(32, 8), 0, stream>>>(qkv, attnA, N, C);
    }
    // 5) proj GEMM (f32 out, bias fused): 128x128 block tiles
    {
        int blocks = (BN / 128) * (C / 128);
        gemm_wmma_kernel<false><<<dim3(blocks), dim3(32, 8), 0, stream>>>(
            attnA, wprojB, b_proj, (void*)out, BN, C, C);
    }
}
